// SelfAttention_11562051960889
// MI455X (gfx1250) — matrix operand order repaired and checked by reading, not compiled, not run on hardware
//
// MI455X (gfx1250) causal self-attention block.
// All matmuls via v_wmma_f32_16x16x32_bf16 (f32 -> bf16 RNE, f32 accum).
// Flash-attention keeps S/P on-chip; K/V tiles are moved by the Tensor Data
// Mover (tensor_load_to_lds, TENSORcnt, hardware row-padding) and are
// double-buffered so DMA overlaps WMMA; Q and out_gemm A tiles use
// global_load_async_to_lds_b128 (ASYNCcnt).
//
// Workspace layout (needs 64 MiB):
//   [0)              q bf16: [b=4][h=16][s=2048][dh=64]
//   [+PLANE)         k bf16: [b=4][h=16][s=2048][dh=64]
//   [+2*PLANE)       v bf16 TRANSPOSED: [b=4][h=16][dh=64][s=2048]
//   [+3*PLANE)       attn bf16: [b*s=8192][1024]  (A of the out-projection)

#include <hip/hip_runtime.h>
#include <stdint.h>

typedef __bf16 bf16_t;
typedef bf16_t v16bf __attribute__((ext_vector_type(16)));
typedef bf16_t v2bf  __attribute__((ext_vector_type(2)));
typedef float  v8f   __attribute__((ext_vector_type(8)));

#define PLANE ((size_t)4 * 16 * 2048 * 64)  // 8388608 elems per q/k/v plane

__device__ __forceinline__ unsigned int pack2(float lo, float hi) {
#if defined(__has_builtin) && __has_builtin(__builtin_amdgcn_cvt_pk_bf16_f32)
  union { v2bf v; unsigned int u; } c;
  c.v = __builtin_amdgcn_cvt_pk_bf16_f32(lo, hi);
  return c.u;
#else
  union { float f; unsigned int u; } a, b;
  a.f = lo; b.f = hi;
  unsigned int ua = a.u + 0x7FFFu + ((a.u >> 16) & 1u);
  unsigned int ub = b.u + 0x7FFFu + ((b.u >> 16) & 1u);
  return (ua >> 16) | (ub & 0xFFFF0000u);
#endif
}
__device__ __forceinline__ unsigned short f32_to_bf16(float f) {
  return (unsigned short)(pack2(f, 0.f) & 0xFFFFu);
}

union Frag { v16bf v; uint4 q[2]; unsigned int u[8]; };
union Acc  { v8f  v; float f[8]; };

// A-matrix 16x32 bf16 fragment: lane holds row (lane&15); elements 0-15 at
// K = 16*g.. (g = lane>>4), contiguous. Two ds_load_b128.
__device__ __forceinline__ v16bf load_afrag(const unsigned short* rowbase, int g) {
  Frag f;
  f.q[0] = *(const uint4*)(rowbase + 16 * g);
  f.q[1] = *(const uint4*)(rowbase + 16 * g + 8);
  return f.v;
}
// B-matrix 32x16 bf16 fragment: lane holds column (lane&15); dword r holds
// K-pair {16g+2r, 16g+2r+1}. colbase points at [col][K=0] with K innermost.
__device__ __forceinline__ v16bf load_bfrag(const unsigned short* colbase, int g) {
  Frag f;
  const uint4* p = (const uint4*)(colbase + 16 * g);
  f.q[0] = p[0];
  f.q[1] = p[1];
  return f.v;
}
__device__ __forceinline__ v8f wmma_bf16(v16bf a, v16bf b, v8f c) {
  return __builtin_amdgcn_wmma_f32_16x16x32_bf16(false, a, false, b,
                                                 (short)0, c, false, false);
}

// CDNA5 async copy: 16 bytes global -> LDS, tracked by ASYNCcnt.
__device__ __forceinline__ void async_b128(unsigned lds_byte_addr, const void* gaddr) {
  asm volatile("global_load_async_to_lds_b128 %0, %1, off"
               :: "v"(lds_byte_addr), "v"(gaddr) : "memory");
}
__device__ __forceinline__ void wait_async0() {
#if defined(__has_builtin) && __has_builtin(__builtin_amdgcn_s_wait_asynccnt)
  __builtin_amdgcn_s_wait_asynccnt(0);
#else
  asm volatile("s_wait_asynccnt 0x0" ::: "memory");
#endif
}
__device__ __forceinline__ void wait_tensor0() {
#if defined(__has_builtin) && __has_builtin(__builtin_amdgcn_s_wait_tensorcnt)
  __builtin_amdgcn_s_wait_tensorcnt(0);
#else
  asm volatile("s_wait_tensorcnt 0x0" ::: "memory");
#endif
}

// Async-copy a 64x64 bf16 tile (global row stride src_stride elems) into LDS
// laid out with 72-elem (144 B) row stride. 128 threads x 4 b128 each.
__device__ __forceinline__ void async_tile_64x64(unsigned short* lds_dst,
                                                 const unsigned short* src,
                                                 size_t src_stride, int t) {
  unsigned lbase = (unsigned)(uintptr_t)lds_dst;  // low 32 bits = LDS offset
#pragma unroll
  for (int p = 0; p < 4; ++p) {
    int idx = t + p * 128;
    int row = idx >> 3, c = idx & 7;  // 8 x 16B per 128B row
    async_b128(lbase + row * 144 + c * 16,
               (const char*)(src + (size_t)row * src_stride) + c * 16);
  }
}

// --- Tensor Data Mover: one-shot 64x64 bf16 tile, global -> LDS, with
// hardware row padding (128B row + 16B pad = 144B LDS stride). D# per ISA §8.
// This toolchain's builtin is the 6-arg form:
//   (u32x4 g0, i32x8 g1, i32x4 g2, i32x4 g3, i32x8, i32 cpol)
#if defined(__has_builtin) && __has_builtin(__builtin_amdgcn_tensor_load_to_lds)
#define HAVE_TDM 1
typedef unsigned int u32x4 __attribute__((ext_vector_type(4)));
typedef int i32x8 __attribute__((ext_vector_type(8)));
typedef int i32x4 __attribute__((ext_vector_type(4)));
__device__ __forceinline__ void tdm_tile_64x64(unsigned lds_byte_addr,
                                               const void* gp,
                                               unsigned row_stride_elems) {
  unsigned long long ga = (unsigned long long)(uintptr_t)gp;
  u32x4 g0;
  g0[0] = 1u;                                   // count=1 (valid user D#)
  g0[1] = lds_byte_addr;                        // lds_addr [63:32]
  g0[2] = (unsigned)ga;                         // global_addr[31:0]
  g0[3] = (unsigned)((ga >> 32) & 0x01FFFFFFu)  // global_addr[56:32]
          | (2u << 30);                         // type=2 ("image")
  i32x8 g1;
  g1[0] = (int)((1u << 16)      // data_size = 2 bytes
                | (1u << 20)    // pad_enable
                | (4u << 22)    // pad_interval: 32 dwords = 128 B
                | (3u << 25));  // pad_amount: 4 dwords = 16 B
  g1[1] = (int)(64u << 16);     // tensor_dim0[15:0]=64 @ bits 63:48
  g1[2] = (int)(64u << 16);     // dim0[31:16]=0 | tensor_dim1[15:0]=64
  g1[3] = (int)(64u << 16);     // dim1[31:16]=0 | tile_dim0=64
  g1[4] = (int)64;              // tile_dim1=64, tile_dim2=0
  g1[5] = (int)row_stride_elems;// tensor_dim0_stride[31:0]
  g1[6] = 0;                    // stride0[47:32]=0 | stride1[15:0]=0
  g1[7] = 0;
  i32x4 z4 = {0, 0, 0, 0};
  i32x8 z8 = {0, 0, 0, 0, 0, 0, 0, 0};
  __builtin_amdgcn_tensor_load_to_lds(g0, g1, z4, z4, z8, 0);
}
#else
#define HAVE_TDM 0
#endif

// ---------------------------------------------------------------------------
// Kernel 1: qkv = x @ W_in + b_in, scattered to bf16 planes (V transposed).
// Block tile 128x64, BK=32, 8 waves (4x2), wave tile 32x32.
// ---------------------------------------------------------------------------
__global__ __launch_bounds__(256) void qkv_gemm(const float* __restrict__ x,
                                                const float* __restrict__ W,
                                                const float* __restrict__ bias,
                                                unsigned short* __restrict__ qkv) {
  const int n0 = blockIdx.x * 64;
  const int m0 = blockIdx.y * 128;
  const int t = threadIdx.x;
  const int lane = t & 31, wave = t >> 5;
  const int lr = lane & 15, g = lane >> 4;
  const int wm = wave >> 1, wn = wave & 1;

  __shared__ __align__(16) unsigned short As[128 * 40];  // [m][k], stride 40
  __shared__ __align__(16) unsigned short Bt[64 * 40];   // [n][k] (transposed)

  Acc acc[2][2];
#pragma unroll
  for (int i = 0; i < 2; ++i)
#pragma unroll
    for (int j = 0; j < 2; ++j)
#pragma unroll
      for (int v = 0; v < 8; ++v) acc[i][j].f[v] = 0.f;

  for (int k0 = 0; k0 < 1024; k0 += 32) {
    __syncthreads();
#pragma unroll
    for (int p = 0; p < 4; ++p) {  // A: 128x32 f32 -> bf16
      int r = (t >> 3) + p * 32;
      int c = (t & 7) * 4;
      float4 f = *(const float4*)(x + (size_t)(m0 + r) * 1024 + k0 + c);
      unsigned int* d = (unsigned int*)(As + r * 40 + c);
      d[0] = pack2(f.x, f.y);
      d[1] = pack2(f.z, f.w);
    }
#pragma unroll
    for (int p = 0; p < 2; ++p) {  // B: 32x64 f32 -> bf16, transposed
      int kr = (t >> 4) + p * 16;
      int c = (t & 15) * 4;
      float4 f = *(const float4*)(W + (size_t)(k0 + kr) * 3072 + n0 + c);
      Bt[(c + 0) * 40 + kr] = f32_to_bf16(f.x);
      Bt[(c + 1) * 40 + kr] = f32_to_bf16(f.y);
      Bt[(c + 2) * 40 + kr] = f32_to_bf16(f.z);
      Bt[(c + 3) * 40 + kr] = f32_to_bf16(f.w);
    }
    __syncthreads();
    v16bf af0 = load_afrag(As + (wm * 32 + 0 + lr) * 40, g);
    v16bf af1 = load_afrag(As + (wm * 32 + 16 + lr) * 40, g);
    v16bf bf0 = load_bfrag(Bt + (wn * 32 + 0 + lr) * 40, g);
    v16bf bf1 = load_bfrag(Bt + (wn * 32 + 16 + lr) * 40, g);
    acc[0][0].v = wmma_bf16(af0, bf0, acc[0][0].v);
    acc[0][1].v = wmma_bf16(af0, bf1, acc[0][1].v);
    acc[1][0].v = wmma_bf16(af1, bf0, acc[1][0].v);
    acc[1][1].v = wmma_bf16(af1, bf1, acc[1][1].v);
  }

  // Branchless scatter. which = n0>>10 is block-uniform; both layouts share
  // the same (b,h) stride 131072. Q/K: [bh][s][dh]; V: [bh][dh][s].
  const int which = n0 >> 10;
  const size_t pbase = (size_t)which * PLANE;
  const size_t sStr = (which == 2) ? 1 : 64;
  const size_t dStr = (which == 2) ? 2048 : 1;
#pragma unroll
  for (int i = 0; i < 2; ++i)
#pragma unroll
    for (int j = 0; j < 2; ++j)
#pragma unroll
      for (int v = 0; v < 8; ++v) {
        int mg = m0 + wm * 32 + i * 16 + v + 8 * g;  // C layout: row v(+8g)
        int ng = n0 + wn * 32 + j * 16 + lr;         // col = lane&15
        float val = acc[i][j].f[v] + bias[ng];
        int rem = ng & 1023;
        int hh = rem >> 6, dd = rem & 63;
        int bb = mg >> 11, ss = mg & 2047;
        size_t dst = pbase + (size_t)(bb * 16 + hh) * 131072 +
                     (size_t)ss * sStr + (size_t)dd * dStr;
        qkv[dst] = f32_to_bf16(val);
      }
}

// ---------------------------------------------------------------------------
// Kernel 2: flash attention per (b,h, 64-row q tile). 4 waves x 16 q rows.
// K/V tiles double-buffered; moved by TDM (wave 0 issues, TENSORcnt waits).
// ---------------------------------------------------------------------------
__global__ __launch_bounds__(128) void flash_attn(
    const unsigned short* __restrict__ qkv, unsigned short* __restrict__ attn) {
  const int qt = blockIdx.x;            // 0..31 (q tile of 64 rows)
  const int bh = blockIdx.y;            // 0..63
  const int batch = bh >> 4, head = bh & 15;
  const int t = threadIdx.x, w = t >> 5, lane = t & 31;
  const int lr = lane & 15, g = lane >> 4;

  __shared__ __align__(16) unsigned short Qs[64 * 72];     // [q][d]
  __shared__ __align__(16) unsigned short Ks[2][64 * 72];  // [k][d]
  __shared__ __align__(16) unsigned short Vt[2][64 * 72];  // [d][k]
  __shared__ __align__(16) unsigned short Ps[4 * 16 * 72]; // per-wave [q][k]

  const unsigned short* qbase = qkv + (size_t)bh * 2048 * 64;
  const unsigned short* kbase = qkv + PLANE + (size_t)bh * 2048 * 64;
  const unsigned short* vtb   = qkv + 2 * PLANE + (size_t)bh * 64 * 2048;  // [d][s]

  const int uw = __builtin_amdgcn_readfirstlane(t) >> 5;  // uniform wave id

  // Q tile via per-lane async; first K/V tile via TDM (one descriptor each).
  async_tile_64x64(Qs, qbase + (size_t)qt * 64 * 64, 64, t);
#if HAVE_TDM
  if (uw == 0) {
    tdm_tile_64x64((unsigned)(uintptr_t)Ks[0], kbase, 64);
    tdm_tile_64x64((unsigned)(uintptr_t)Vt[0], vtb, 2048);
  }
#else
  async_tile_64x64(Ks[0], kbase, 64, t);
  async_tile_64x64(Vt[0], vtb, 2048, t);
#endif
  wait_async0();
#if HAVE_TDM
  if (uw == 0) wait_tensor0();
#endif
  __syncthreads();

  v16bf qf0 = load_afrag(Qs + (w * 16 + lr) * 72 + 0, g);   // d 0..31
  v16bf qf1 = load_afrag(Qs + (w * 16 + lr) * 72 + 32, g);  // d 32..63

  Acc o[4];
  float ms[8], ls[8];
#pragma unroll
  for (int n = 0; n < 4; ++n)
#pragma unroll
    for (int v = 0; v < 8; ++v) o[n].f[v] = 0.f;
#pragma unroll
  for (int v = 0; v < 8; ++v) { ms[v] = -__builtin_inff(); ls[v] = 0.f; }

  for (int kt = 0; kt <= qt; ++kt) {
    const int cur = kt & 1;
    if (kt < qt) {  // prefetch next tile into the other buffer (DMA || WMMA)
#if HAVE_TDM
      if (uw == 0) {
        tdm_tile_64x64((unsigned)(uintptr_t)Ks[1 - cur],
                       kbase + (size_t)(kt + 1) * 64 * 64, 64);
        tdm_tile_64x64((unsigned)(uintptr_t)Vt[1 - cur],
                       vtb + (size_t)(kt + 1) * 64, 2048);
      }
#else
      async_tile_64x64(Ks[1 - cur], kbase + (size_t)(kt + 1) * 64 * 64, 64, t);
      async_tile_64x64(Vt[1 - cur], vtb + (size_t)(kt + 1) * 64, 2048, t);
#endif
    }

    // S = Q K^T  (16 q rows x 64 k cols per wave)
    Acc s[4];
#pragma unroll
    for (int n = 0; n < 4; ++n)
#pragma unroll
      for (int v = 0; v < 8; ++v) s[n].f[v] = 0.f;
#pragma unroll
    for (int n = 0; n < 4; ++n) {
      v16bf kf0 = load_bfrag(Ks[cur] + (n * 16 + lr) * 72 + 0, g);
      v16bf kf1 = load_bfrag(Ks[cur] + (n * 16 + lr) * 72 + 32, g);
      s[n].v = wmma_bf16(qf0, kf0, s[n].v);
      s[n].v = wmma_bf16(qf1, kf1, s[n].v);
    }

    const bool diag = (kt == qt);
#pragma unroll
    for (int v = 0; v < 8; ++v) {
      int rloc = w * 16 + v + 8 * g;  // row within the 64-row q tile
      float tm = -__builtin_inff();
      float vals[4];
#pragma unroll
      for (int n = 0; n < 4; ++n) {
        float xv = s[n].f[v] * 0.125f;  // 1/sqrt(64)
        if (diag && (n * 16 + lr) > rloc) xv = -__builtin_inff();
        vals[n] = xv;
        tm = fmaxf(tm, xv);
      }
#pragma unroll
      for (int off = 1; off < 16; off <<= 1) tm = fmaxf(tm, __shfl_xor(tm, off, 32));
      float mnew = fmaxf(ms[v], tm);
      float alpha = __expf(ms[v] - mnew);
      float rs = 0.f;
#pragma unroll
      for (int n = 0; n < 4; ++n) {
        float pv = __expf(vals[n] - mnew);
        s[n].f[v] = pv;
        rs += pv;
      }
#pragma unroll
      for (int off = 1; off < 16; off <<= 1) rs += __shfl_xor(rs, off, 32);
      ls[v] = ls[v] * alpha + rs;
      ms[v] = mnew;
#pragma unroll
      for (int n = 0; n < 4; ++n) o[n].f[v] *= alpha;
    }

    // P (C layout) -> per-wave LDS [16][64] bf16 (wave-private; DS in-order)
    unsigned short* pp = Ps + w * (16 * 72);
#pragma unroll
    for (int v = 0; v < 8; ++v)
#pragma unroll
      for (int n = 0; n < 4; ++n)
        pp[(v + 8 * g) * 72 + n * 16 + lr] = f32_to_bf16(s[n].f[v]);

    // O += P @ V
#pragma unroll
    for (int ks = 0; ks < 2; ++ks) {
      v16bf pf = load_afrag(pp + lr * 72 + ks * 32, g);
#pragma unroll
      for (int n = 0; n < 4; ++n) {
        v16bf vf = load_bfrag(Vt[cur] + (n * 16 + lr) * 72 + ks * 32, g);
        o[n].v = wmma_bf16(pf, vf, o[n].v);
      }
    }

    // next tile landed + every wave done reading buffer cur
#if HAVE_TDM
    if (uw == 0) wait_tensor0();
#else
    wait_async0();
#endif
    __syncthreads();
  }

#pragma unroll
  for (int v = 0; v < 8; ++v) {
    float inv = 1.0f / ls[v];
    int qg = qt * 64 + w * 16 + v + 8 * g;
    size_t rowoff = ((size_t)(batch * 2048 + qg)) * 1024 + head * 64;
#pragma unroll
    for (int n = 0; n < 4; ++n)
      attn[rowoff + n * 16 + lr] = f32_to_bf16(o[n].f[v] * inv);
  }
}

// ---------------------------------------------------------------------------
// Kernel 3: out = attn(bf16) @ W_out + b_out  (f32 result).
// A tile streams into LDS with async loads, overlapped with B conversion.
// ---------------------------------------------------------------------------
__global__ __launch_bounds__(256) void out_gemm(
    const unsigned short* __restrict__ attn, const float* __restrict__ W,
    const float* __restrict__ bias, float* __restrict__ out) {
  const int n0 = blockIdx.x * 64;
  const int m0 = blockIdx.y * 128;
  const int t = threadIdx.x;
  const int lane = t & 31, wave = t >> 5;
  const int lr = lane & 15, g = lane >> 4;
  const int wm = wave >> 1, wn = wave & 1;

  __shared__ __align__(16) unsigned short As[128 * 40];
  __shared__ __align__(16) unsigned short Bt[64 * 40];

  Acc acc[2][2];
#pragma unroll
  for (int i = 0; i < 2; ++i)
#pragma unroll
    for (int j = 0; j < 2; ++j)
#pragma unroll
      for (int v = 0; v < 8; ++v) acc[i][j].f[v] = 0.f;

  const unsigned lAs = (unsigned)(uintptr_t)As;
  for (int k0 = 0; k0 < 1024; k0 += 32) {
    __syncthreads();
    // A tile 128x32 bf16: async copy (512 x b128 / 256 threads = 2 each)
#pragma unroll
    for (int p = 0; p < 2; ++p) {
      int idx = t + p * 256;
      int row = idx >> 2, c = idx & 3;  // 4 x 16B per 64B row
      async_b128(lAs + row * 80 + c * 16,
                 (const char*)(attn + (size_t)(m0 + row) * 1024 + k0) + c * 16);
    }
#pragma unroll
    for (int p = 0; p < 2; ++p) {  // B tile 32x64 f32 -> bf16 transposed
      int kr = (t >> 4) + p * 16;
      int c = (t & 15) * 4;
      float4 f = *(const float4*)(W + (size_t)(k0 + kr) * 1024 + n0 + c);
      Bt[(c + 0) * 40 + kr] = f32_to_bf16(f.x);
      Bt[(c + 1) * 40 + kr] = f32_to_bf16(f.y);
      Bt[(c + 2) * 40 + kr] = f32_to_bf16(f.z);
      Bt[(c + 3) * 40 + kr] = f32_to_bf16(f.w);
    }
    wait_async0();
    __syncthreads();
    v16bf af0 = load_afrag(As + (wm * 32 + 0 + lr) * 40, g);
    v16bf af1 = load_afrag(As + (wm * 32 + 16 + lr) * 40, g);
    v16bf bf0 = load_bfrag(Bt + (wn * 32 + 0 + lr) * 40, g);
    v16bf bf1 = load_bfrag(Bt + (wn * 32 + 16 + lr) * 40, g);
    acc[0][0].v = wmma_bf16(af0, bf0, acc[0][0].v);
    acc[0][1].v = wmma_bf16(af0, bf1, acc[0][1].v);
    acc[1][0].v = wmma_bf16(af1, bf0, acc[1][0].v);
    acc[1][1].v = wmma_bf16(af1, bf1, acc[1][1].v);
  }
#pragma unroll
  for (int i = 0; i < 2; ++i)
#pragma unroll
    for (int j = 0; j < 2; ++j)
#pragma unroll
      for (int v = 0; v < 8; ++v) {
        int mg = m0 + wm * 32 + i * 16 + v + 8 * g;
        int ng = n0 + wn * 32 + j * 16 + lr;
        out[(size_t)mg * 1024 + ng] = acc[i][j].f[v] + bias[ng];
      }
}

extern "C" void kernel_launch(void* const* d_in, const int* in_sizes, int n_in,
                              void* d_out, int out_size, void* d_ws, size_t ws_size,
                              hipStream_t stream) {
  const float* x     = (const float*)d_in[0];  // [4,2048,1024]
  const float* W_in  = (const float*)d_in[1];  // [1024,3072]
  const float* b_in  = (const float*)d_in[2];  // [3072]
  const float* W_out = (const float*)d_in[3];  // [1024,1024]
  const float* b_out = (const float*)d_in[4];  // [1024]
  float* out = (float*)d_out;                  // [4,2048,1024]

  const size_t need = 3 * PLANE * 2 + (size_t)8192 * 1024 * 2;  // 64 MiB
  if (ws_size < need) return;
  unsigned short* qkv  = (unsigned short*)d_ws;
  unsigned short* attn = qkv + 3 * PLANE;

  dim3 g1(3072 / 64, 8192 / 128);  // (48, 64)
  qkv_gemm<<<g1, 256, 0, stream>>>(x, W_in, b_in, qkv);

  dim3 g2(2048 / 64, 4 * 16);      // (32, 64)
  flash_attn<<<g2, 128, 0, stream>>>(qkv, attn);

  dim3 g3(1024 / 64, 8192 / 128);  // (16, 64)
  out_gemm<<<g3, 256, 0, stream>>>(attn, W_out, b_out, out);
}
